// DenseFlashAttention_23063974379958
// MI455X (gfx1250) — compile-verified, matrix operand order checked by reading, not run on hardware
//
#include <hip/hip_runtime.h>
#include <hip/hip_bf16.h>

#define FEAT  64
#define HEADS 4

typedef __attribute__((ext_vector_type(16))) __bf16 v16bf;
typedef __attribute__((ext_vector_type(8)))  float  v8f;
typedef __attribute__((ext_vector_type(8)))  __bf16 bf16x8;

// ---------------------------------------------------------------------------
// Init workspace: agg = 0, denom = 0, segment-max = -inf
// ---------------------------------------------------------------------------
__global__ void gat_init_kernel(float* __restrict__ mbuf,
                                float* __restrict__ denom,
                                float* __restrict__ agg, int N) {
    size_t idx = (size_t)blockIdx.x * blockDim.x + threadIdx.x;
    size_t total = (size_t)HEADS * N * FEAT;
    if (idx < total) agg[idx] = 0.0f;
    if (idx < (size_t)N * HEADS) {
        mbuf[idx]  = -__builtin_inff();
        denom[idx] = 0.0f;
    }
}

// ---------------------------------------------------------------------------
// WMMA fragment-slot mapping (inverse of the 16-bit A/B 16x32 VGPR layout,
// ISA 7.12.2).  Element (m-or-n = m, k) lives at [m>>4][k>>5][lane][j].
//   k%32 = hi16*16 + lhalf*8 + (v&3)*2 + parity ;  lane = lhalf*16 + (m&15)
//   j = 2*(hi16*4 + (v&3)) + parity
// ---------------------------------------------------------------------------
__device__ __forceinline__ void frag_slot(int m, int k,
                                          int& part, int& ks, int& lane, int& j) {
    part = m >> 4;
    ks   = k >> 5;
    int k2    = k & 31;
    int hi    = (k2 >> 4) & 1;
    int lhalf = (k2 >> 3) & 1;
    int pos   = k2 & 7;
    lane = lhalf * 16 + (m & 15);
    j    = ((hi << 2) | (pos >> 1)) * 2 + (pos & 1);
}

__device__ __forceinline__ v8f wmma_bf16(v16bf a, v16bf b, v8f acc) {
    return __builtin_amdgcn_wmma_f32_16x16x32_bf16(
        false, a, false, b, (short)0, acc, false, false);
}

// ---------------------------------------------------------------------------
// QKV projection: per (64-node tile, head, {Q,K,V}) block, 256 threads.
// x,W staged into LDS in WMMA-fragment order -> each fragment is one aligned
// 32B per-lane read (2x ds_load_b128).  D goes through LDS for coalesced
// 16B global stores.
// ---------------------------------------------------------------------------
__global__ void gat_qkv_kernel(const float* __restrict__ x,
                               const float* __restrict__ Wq,
                               const float* __restrict__ Wk,
                               const float* __restrict__ Wv,
                               __bf16* __restrict__ Q,
                               __bf16* __restrict__ K,
                               __bf16* __restrict__ V, int N) {
    const int tile  = blockIdx.x;
    const int h     = blockIdx.y;
    const int which = blockIdx.z;
    const float* W = (which == 0 ? Wq : which == 1 ? Wk : Wv) + (size_t)h * FEAT * FEAT;
    __bf16* Out    = (which == 0 ? Q  : which == 1 ? K  : V)  + (size_t)h * N * FEAT;

    __shared__ __align__(32) __bf16 afrag[4][2][32][16];  // [row tile][ks][lane][j]
    __shared__ __align__(32) __bf16 bfrag[4][2][32][16];  // [col tile][ks][lane][j]
    __shared__ __align__(16) __bf16 outs[64][FEAT];

    const int tid   = threadIdx.x;
    const int node0 = tile * 64;

    for (int i = tid; i < 64 * FEAT; i += 256) {
        const int r = i >> 6, c = i & 63;
        const int n = node0 + r;
        const float xv = (n < N) ? x[(size_t)n * FEAT + c] : 0.0f;
        int p, ks, la, j;
        frag_slot(r, c, p, ks, la, j);          // A: m=node row, k=feat
        afrag[p][ks][la][j] = (__bf16)xv;
        frag_slot(c, r, p, ks, la, j);          // B: n=out-feat col, k=in-feat row
        bfrag[p][ks][la][j] = (__bf16)W[i];
    }
    __syncthreads();

    const int wave = tid >> 5, lane = tid & 31;
    const int lrow = lane & 15, lhalf = lane >> 4;
#pragma unroll
    for (int t = wave * 2; t < wave * 2 + 2; ++t) {
        const int tm = t >> 2, tn = t & 3;
        v8f acc = {};
#pragma unroll
        for (int ks = 0; ks < 2; ++ks) {
            v16bf a = *(const v16bf*)(&afrag[tm][ks][lane][0]);
            v16bf b = *(const v16bf*)(&bfrag[tn][ks][lane][0]);
            acc = wmma_bf16(a, b, acc);
        }
#pragma unroll
        for (int vg = 0; vg < 8; ++vg)
            outs[tm * 16 + lhalf * 8 + vg][tn * 16 + lrow] = (__bf16)acc[vg];
    }
    __syncthreads();

    // Coalesced 16B stores (8 bf16 per chunk), one tail guard per chunk.
    for (int i = tid; i < 64 * (FEAT / 8); i += 256) {
        const int r = i >> 3, c8 = (i & 7) * 8;
        const int node = node0 + r;
        if (node < N)
            *(uint4*)(Out + (size_t)node * FEAT + c8) = *(const uint4*)(&outs[r][c8]);
    }
}

// ---------------------------------------------------------------------------
// Edge scores: s[e,h] = scale * dot(Q[h,recv], K[h,send]); atomic segment max.
// ---------------------------------------------------------------------------
__global__ void gat_edge_score_kernel(const int* __restrict__ ei,
                                      const __bf16* __restrict__ Q,
                                      const __bf16* __restrict__ K,
                                      float* __restrict__ s,
                                      float* __restrict__ mbuf, int N, int E) {
    int idx = blockIdx.x * blockDim.x + threadIdx.x;
    if (idx >= E * HEADS) return;
    const int e = idx >> 2, h = idx & 3;
    const int snd = ei[e], rcv = ei[E + e];
    const bf16x8* q = (const bf16x8*)(Q + ((size_t)h * N + rcv) * FEAT);
    const bf16x8* k = (const bf16x8*)(K + ((size_t)h * N + snd) * FEAT);
    float acc = 0.0f;
#pragma unroll
    for (int i = 0; i < 8; ++i) {
        bf16x8 qv = q[i], kv = k[i];
#pragma unroll
        for (int j = 0; j < 8; ++j) acc += (float)qv[j] * (float)kv[j];
    }
    acc *= 0.125f;  // FEAT^-0.5
    s[idx] = acc;
    float* addr = mbuf + (size_t)rcv * HEADS + h;
    if (acc >= 0.0f)
        atomicMax((int*)addr, __float_as_int(acc));
    else
        atomicMin((unsigned int*)addr, __float_as_uint(acc));
}

// ---------------------------------------------------------------------------
// p = exp(s - m[recv]); atomic denom accumulate.  Prefetch the V row the
// aggregate pass will gather (global_prefetch_b8 -> warm L2/L0).
// ---------------------------------------------------------------------------
__global__ void gat_edge_exp_kernel(const int* __restrict__ ei,
                                    float* __restrict__ s,
                                    const float* __restrict__ mbuf,
                                    float* __restrict__ denom,
                                    const __bf16* __restrict__ V,
                                    int N, int E) {
    int idx = blockIdx.x * blockDim.x + threadIdx.x;
    if (idx >= E * HEADS) return;
    const int e = idx >> 2, h = idx & 3;
    const int snd = ei[e], rcv = ei[E + e];
    __builtin_prefetch(V + ((size_t)h * N + snd) * FEAT, 0, 1);
    float p = __expf(s[idx] - mbuf[(size_t)rcv * HEADS + h]);
    s[idx] = p;
    atomicAdd(denom + (size_t)rcv * HEADS + h, p);
}

// ---------------------------------------------------------------------------
// Aggregate: one wave32 per edge; lane f handles features {f, f+32} for all
// 4 heads -> coalesced global_atomic_add_f32 into agg[h, recv, :].
// ---------------------------------------------------------------------------
__global__ void gat_edge_agg_kernel(const int* __restrict__ ei,
                                    const float* __restrict__ p,
                                    const float* __restrict__ denom,
                                    const __bf16* __restrict__ V,
                                    float* __restrict__ agg, int N, int E) {
    int gid  = blockIdx.x * blockDim.x + threadIdx.x;
    int e    = gid >> 5;
    int lane = gid & 31;
    if (e >= E) return;
    const int snd = ei[e], rcv = ei[E + e];
#pragma unroll
    for (int h = 0; h < HEADS; ++h) {
        float alpha = p[(size_t)e * HEADS + h] / denom[(size_t)rcv * HEADS + h];
        const __bf16* vrow = V + ((size_t)h * N + snd) * FEAT;
        float* arow = agg + ((size_t)h * N + rcv) * FEAT;
        atomicAdd(arow + lane,      alpha * (float)vrow[lane]);
        atomicAdd(arow + lane + 32, alpha * (float)vrow[lane + 32]);
    }
}

// ---------------------------------------------------------------------------
// Output: out = x + mean_h(agg) @ Wout   (WMMA bf16, fp32 residual, float4 IO)
// ---------------------------------------------------------------------------
__global__ void gat_out_kernel(const float* __restrict__ x,
                               const float* __restrict__ Wout,
                               const float* __restrict__ agg,
                               float* __restrict__ out, int N) {
    __shared__ __align__(32) __bf16 afrag[4][2][32][16];
    __shared__ __align__(32) __bf16 bfrag[4][2][32][16];
    __shared__ __align__(16) float  outs[64][FEAT];

    const int tid   = threadIdx.x;
    const int node0 = blockIdx.x * 64;

    for (int i = tid; i < 64 * FEAT; i += 256) {
        const int r = i >> 6, c = i & 63;
        const int n = node0 + r;
        float v = 0.0f;
        if (n < N) {
#pragma unroll
            for (int h = 0; h < HEADS; ++h) v += agg[((size_t)h * N + n) * FEAT + c];
            v *= 0.25f;  // mean over heads (deg-0 nodes stay exactly 0)
        }
        int p, ks, la, j;
        frag_slot(r, c, p, ks, la, j);
        afrag[p][ks][la][j] = (__bf16)v;
        frag_slot(c, r, p, ks, la, j);
        bfrag[p][ks][la][j] = (__bf16)Wout[i];
    }
    __syncthreads();

    const int wave = tid >> 5, lane = tid & 31;
    const int lrow = lane & 15, lhalf = lane >> 4;
#pragma unroll
    for (int t = wave * 2; t < wave * 2 + 2; ++t) {
        const int tm = t >> 2, tn = t & 3;
        v8f acc = {};
#pragma unroll
        for (int ks = 0; ks < 2; ++ks) {
            v16bf a = *(const v16bf*)(&afrag[tm][ks][lane][0]);
            v16bf b = *(const v16bf*)(&bfrag[tn][ks][lane][0]);
            acc = wmma_bf16(a, b, acc);
        }
#pragma unroll
        for (int vg = 0; vg < 8; ++vg)
            outs[tm * 16 + lhalf * 8 + vg][tn * 16 + lrow] = acc[vg];
    }
    __syncthreads();

    // Residual add + coalesced float4 stores.
    for (int i = tid; i < 64 * (FEAT / 4); i += 256) {
        const int r = i >> 4, c4 = (i & 15) * 4;
        const int node = node0 + r;
        if (node < N) {
            const size_t o = (size_t)node * FEAT + c4;
            float4 xv = *(const float4*)(x + o);
            float4 mv = *(const float4*)(&outs[r][c4]);
            float4 res = make_float4(xv.x + mv.x, xv.y + mv.y,
                                     xv.z + mv.z, xv.w + mv.w);
            *(float4*)(out + o) = res;
        }
    }
}

// ---------------------------------------------------------------------------
extern "C" void kernel_launch(void* const* d_in, const int* in_sizes, int n_in,
                              void* d_out, int out_size, void* d_ws, size_t ws_size,
                              hipStream_t stream) {
    const float* x    = (const float*)d_in[0];
    const float* Wq   = (const float*)d_in[1];
    const float* Wk   = (const float*)d_in[2];
    const float* Wv   = (const float*)d_in[3];
    const float* Wout = (const float*)d_in[4];
    const int*   ei   = (const int*)d_in[5];

    const int N = in_sizes[0] / FEAT;
    const int E = in_sizes[5] / 2;

    // Workspace carve (bf16 QKV 76.8MB + fp32 agg 51.2MB + scores 6.4MB +
    // max/denom 1.6MB ~= 136MB -> resident in the 192MB L2).
    size_t off = 0;
    auto carve = [&](size_t bytes) -> void* {
        void* p = (char*)d_ws + off;
        off += (bytes + 255) & ~(size_t)255;
        return p;
    };
    __bf16* Q     = (__bf16*)carve((size_t)HEADS * N * FEAT * sizeof(__bf16));
    __bf16* K     = (__bf16*)carve((size_t)HEADS * N * FEAT * sizeof(__bf16));
    __bf16* V     = (__bf16*)carve((size_t)HEADS * N * FEAT * sizeof(__bf16));
    float*  s     = (float*)carve((size_t)E * HEADS * sizeof(float));
    float*  mbuf  = (float*)carve((size_t)N * HEADS * sizeof(float));
    float*  denom = (float*)carve((size_t)N * HEADS * sizeof(float));
    float*  agg   = (float*)carve((size_t)HEADS * N * FEAT * sizeof(float));

    const size_t init_n = (size_t)HEADS * N * FEAT;
    gat_init_kernel<<<(unsigned)((init_n + 255) / 256), 256, 0, stream>>>(mbuf, denom, agg, N);

    dim3 qkv_grid((N + 63) / 64, HEADS, 3);
    gat_qkv_kernel<<<qkv_grid, 256, 0, stream>>>(x, Wq, Wk, Wv, Q, K, V, N);

    const int eh = E * HEADS;
    gat_edge_score_kernel<<<(eh + 255) / 256, 256, 0, stream>>>(ei, Q, K, s, mbuf, N, E);
    gat_edge_exp_kernel<<<(eh + 255) / 256, 256, 0, stream>>>(ei, s, mbuf, denom, V, N, E);

    const long long agg_threads = (long long)E * 32;
    gat_edge_agg_kernel<<<(unsigned)((agg_threads + 255) / 256), 256, 0, stream>>>(
        ei, s, denom, V, agg, N, E);

    gat_out_kernel<<<(N + 63) / 64, 256, 0, stream>>>(x, Wout, agg, (float*)d_out, N);
}
